// Attention_7481833030249
// MI455X (gfx1250) — matrix operand order repaired and checked by reading, not compiled, not run on hardware
//
#include <hip/hip_runtime.h>
#include <hip/hip_bf16.h>

typedef __attribute__((ext_vector_type(16))) __bf16        v16bf;
typedef __attribute__((ext_vector_type(8)))  float         v8f;
typedef __attribute__((ext_vector_type(8)))  unsigned int  v8u;

#define B_       32
#define N_       577
#define DIM_     768
#define H_       12
#define DH_      64
#define M_TOK    (B_ * N_)        // 18464
#define VST_     640              // padded V^T row stride (>= 577+63, 16B aligned)
#define NEG_     (-987654321.0f)
#define NINF_    (-3.0e38f)

// ---- gfx1250 async global->LDS path (guarded; falls back to sync copy) ----
#if defined(__HIP_DEVICE_COMPILE__) && defined(__gfx1250__) && \
    __has_builtin(__builtin_amdgcn_global_load_async_to_lds_b128)
#define ASYNC_LDS 1
#else
#define ASYNC_LDS 0
#endif

#if ASYNC_LDS
typedef int a4i __attribute__((vector_size(16)));
typedef __attribute__((address_space(1))) a4i* gp4_t;   // global v4i*
typedef __attribute__((address_space(3))) a4i* lp4_t;   // LDS v4i*
#endif

__device__ __forceinline__ void async_copy16(const unsigned short* g, unsigned short* l) {
#if ASYNC_LDS
  __builtin_amdgcn_global_load_async_to_lds_b128((gp4_t)g, (lp4_t)l, 0, 0);
#else
  *(uint4*)l = *(const uint4*)g;
#endif
}
__device__ __forceinline__ void async_wait() {
#if ASYNC_LDS
#if __has_builtin(__builtin_amdgcn_s_wait_asynccnt)
  __builtin_amdgcn_s_wait_asynccnt(0);
#else
  asm volatile("s_wait_asynccnt 0x0" ::: "memory");
#endif
#endif
}

__device__ __forceinline__ unsigned short f2bf(float f) {
  unsigned int u = __builtin_bit_cast(unsigned int, f);
  u += 0x7fffu + ((u >> 16) & 1u);          // round-to-nearest-even
  return (unsigned short)(u >> 16);
}

__device__ __forceinline__ v8f v8f_zero() {
  v8f z = {0.f, 0.f, 0.f, 0.f, 0.f, 0.f, 0.f, 0.f};
  return z;
}

// Build a 16-element bf16 fragment from two 16-byte chunks (LDS).
__device__ __forceinline__ v16bf frag16(const unsigned short* p0, const unsigned short* p1) {
  uint4 a = *(const uint4*)p0;
  uint4 b = *(const uint4*)p1;
  v8u u;
  u[0] = a.x; u[1] = a.y; u[2] = a.z; u[3] = a.w;
  u[4] = b.x; u[5] = b.y; u[6] = b.z; u[7] = b.w;
  return __builtin_bit_cast(v16bf, u);
}

#define WMMA_BF16(A, Bf, Cf) \
  __builtin_amdgcn_wmma_f32_16x16x32_bf16(false, (A), false, (Bf), (short)0, (Cf), false, false)

// ---------------------------------------------------------------- converts
__global__ void cvt_f32_bf16(const float* __restrict__ src,
                             unsigned short* __restrict__ dst, int n) {
  int i = blockIdx.x * 256 + threadIdx.x;
  if (i < n) dst[i] = f2bf(src[i]);
}

// w: [K][N] fp32 -> wt: [N][K] bf16 (K contiguous)
__global__ void transpose_w(const float* __restrict__ w,
                            unsigned short* __restrict__ wt, int K, int N) {
  int i = blockIdx.x * 256 + threadIdx.x;
  if (i < K * N) {
    int n = i / K, k = i - n * K;
    wt[i] = f2bf(w[(size_t)k * N + n]);
  }
}

// ---------------------------------------------------------------- GEMM core
// A: [M][K] bf16 row-major.  Bt: [Ncols][K] bf16 row-major (i.e. B transposed).
// 256 threads = 8 waves, block tile 128x128, wave tile 32x64, K step 32.
// Double-buffered LDS filled via async global->LDS DMA.
__device__ __forceinline__ void gemm_mainloop(const unsigned short* __restrict__ A,
                                              const unsigned short* __restrict__ Bt,
                                              int M, int K, int mBase, int nBase,
                                              unsigned short* ldsA, unsigned short* ldsB,
                                              v8f C[2][4]) {
  const int tid  = threadIdx.x;
  const int wave = tid >> 5, lane = tid & 31;
  const int lo   = lane & 15, hi = lane >> 4;
  const int wm   = wave >> 1, wn = wave & 1;
  const int LS   = 40;                           // padded LDS row stride (halves)
  const int BUFS = 128 * LS;                     // per-buffer size (halves)

  for (int i = 0; i < 2; ++i)
    for (int j = 0; j < 4; ++j) C[i][j] = v8f_zero();

  const int r = tid >> 1;
  const int c = (tid & 1) * 16;
  int arow = mBase + r; if (arow >= M) arow = M - 1;
  const unsigned short* gA = A  + (size_t)arow * K + c;
  const unsigned short* gB = Bt + (size_t)(nBase + r) * K + c;

  // preload tile 0 into buffer 0
  async_copy16(gA, ldsA + r * LS + c);  async_copy16(gA + 8, ldsA + r * LS + c + 8);
  async_copy16(gB, ldsB + r * LS + c);  async_copy16(gB + 8, ldsB + r * LS + c + 8);
  async_wait();
  __syncthreads();

  for (int k0 = 0; k0 < K; k0 += 32) {
    const int cur = (k0 >> 5) & 1;
    if (k0 + 32 < K) {                           // async prefetch of next tile
      unsigned short* la = ldsA + (cur ^ 1) * BUFS + r * LS + c;
      unsigned short* lb = ldsB + (cur ^ 1) * BUFS + r * LS + c;
      async_copy16(gA + k0 + 32, la);  async_copy16(gA + k0 + 40, la + 8);
      async_copy16(gB + k0 + 32, lb);  async_copy16(gB + k0 + 40, lb + 8);
      if (k0 + 64 < K) {                         // gfx1250 global_prefetch_b8
        __builtin_prefetch(gA + k0 + 64, 0, 1);
        __builtin_prefetch(gB + k0 + 64, 0, 1);
      }
    }
    const unsigned short* bA = ldsA + cur * BUFS;
    const unsigned short* bB = ldsB + cur * BUFS;
    v16bf afr[2], bfr[4];
    for (int i = 0; i < 2; ++i) {
      const unsigned short* base = bA + (wm * 32 + i * 16 + lo) * LS;
      afr[i] = frag16(base + hi * 8, base + 16 + hi * 8);
    }
    for (int j = 0; j < 4; ++j) {
      const unsigned short* base = bB + (wn * 64 + j * 16 + lo) * LS;
      bfr[j] = frag16(base + hi * 8, base + 16 + hi * 8);
    }
    for (int i = 0; i < 2; ++i)
      for (int j = 0; j < 4; ++j)
        C[i][j] = WMMA_BF16(afr[i], bfr[j], C[i][j]);
    if (k0 + 32 < K) async_wait();
    __syncthreads();
  }
}

// ---------------------------------------------------------------- qkv GEMM
__global__ void __launch_bounds__(256) qkv_gemm(const unsigned short* __restrict__ xb,
                                                const unsigned short* __restrict__ wqt,
                                                const float* __restrict__ scale,
                                                unsigned short* __restrict__ qb,
                                                unsigned short* __restrict__ kb,
                                                unsigned short* __restrict__ vtb) {
  __shared__ unsigned short ldsA[2 * 128 * 40];
  __shared__ unsigned short ldsB[2 * 128 * 40];
  v8f C[2][4];
  const int mBase = blockIdx.x * 128, nBase = blockIdx.y * 128;
  gemm_mainloop(xb, wqt, M_TOK, DIM_, mBase, nBase, ldsA, ldsB, C);

  const int tid = threadIdx.x, wave = tid >> 5, lane = tid & 31;
  const int lo = lane & 15, hi = lane >> 4;
  const int wm = wave >> 1, wn = wave & 1;
  for (int i = 0; i < 2; ++i) {
    int row0 = mBase + wm * 32 + i * 16 + hi * 8;
    for (int j = 0; j < 4; ++j) {
      int col = nBase + wn * 64 + j * 16 + lo;   // [0, 2304)
      int sec = col / DIM_, w = col - sec * DIM_;
      int h = w >> 6, d = w & 63;
      float sc = (sec == 0) ? scale[h] : 1.0f;   // fold LSA scale into Q
      for (int r = 0; r < 8; ++r) {
        int row = row0 + r;
        if (row < M_TOK) {
          int b = row / N_, n = row - b * N_;
          unsigned short bv = f2bf(C[i][j][r] * sc);
          size_t bh = (size_t)(b * H_ + h);
          if (sec == 0)      qb [(bh * N_  + n) * DH_  + d] = bv;  // [B,H,N,DH]
          else if (sec == 1) kb [(bh * N_  + n) * DH_  + d] = bv;  // [B,H,N,DH]
          else               vtb[(bh * DH_ + d) * VST_ + n] = bv;  // [B,H,DH,VST]
        }
      }
    }
  }
}

// ---------------------------------------------------------------- flash attention
// 128 threads = 4 waves; each wave owns 32 query rows; key tiles of 64,
// double-buffered via async global->LDS DMA.
__global__ void __launch_bounds__(128) flash_attn(const unsigned short* __restrict__ qb,
                                                  const unsigned short* __restrict__ kb,
                                                  const unsigned short* __restrict__ vtb,
                                                  unsigned short* __restrict__ aob) {
  __shared__ unsigned short ldsK[2 * 64 * 72];   // [j][d], padded
  __shared__ unsigned short ldsV[2 * 64 * 72];   // [d][j], padded
  __shared__ unsigned short ldsP[4 * 32 * 72];   // per-wave P tile
  const int KT = 64 * 72;

  const int tid = threadIdx.x, wave = tid >> 5, lane = tid & 31;
  const int lo = lane & 15, hi = lane >> 4;
  const int bh = blockIdx.y;                     // b*H + h
  const int b = bh / H_, h = bh - b * H_;
  const int qbase = blockIdx.x * 128 + wave * 32;

  const unsigned short* qrows = qb  + (size_t)bh * N_  * DH_;
  const unsigned short* krows = kb  + (size_t)bh * N_  * DH_;
  const unsigned short* vrows = vtb + (size_t)bh * DH_ * VST_;

  // Q fragments in registers (scale already folded in)
  v16bf qfr[2][2];
  for (int i = 0; i < 2; ++i) {
    int row = qbase + i * 16 + lo; if (row >= N_) row = N_ - 1;
    const unsigned short* p = qrows + (size_t)row * DH_;
    for (int kc = 0; kc < 2; ++kc)
      qfr[i][kc] = frag16(p + kc * 32 + hi * 8, p + kc * 32 + 16 + hi * 8);
  }

  v8f O[2][4];
  for (int i = 0; i < 2; ++i) for (int dt = 0; dt < 4; ++dt) O[i][dt] = v8f_zero();
  float m[2][8], l[2][8];
  for (int i = 0; i < 2; ++i)
    for (int r = 0; r < 8; ++r) { m[i][r] = NINF_; l[i][r] = 0.f; }

  const int rr = tid >> 1, cc = (tid & 1) * 32;  // cooperative tile-load coords
  auto load_tiles = [&](int kt, int bufi) {
    const int kvbase = kt * 64;
    int krow = kvbase + rr; if (krow >= N_) krow = N_ - 1;
    const unsigned short* gk = krows + (size_t)krow * DH_ + cc;
    const unsigned short* gv = vrows + (size_t)rr * VST_ + kvbase + cc;
    unsigned short* lk = ldsK + bufi * KT + rr * 72 + cc;
    unsigned short* lv = ldsV + bufi * KT + rr * 72 + cc;
    for (int t = 0; t < 32; t += 8) {
      async_copy16(gk + t, lk + t);
      async_copy16(gv + t, lv + t);
    }
  };

  load_tiles(0, 0);
  async_wait();
  __syncthreads();

  for (int kt = 0; kt < 10; ++kt) {              // ceil(577/64)
    const int kvbase = kt * 64;
    const int buf = kt & 1;
    if (kt + 1 < 10) load_tiles(kt + 1, buf ^ 1);
    const unsigned short* bKlds = ldsK + buf * KT;
    const unsigned short* bVlds = ldsV + buf * KT;

    // S = Q K^T
    v16bf bK[4][2];
    for (int j = 0; j < 4; ++j) {
      const unsigned short* base = bKlds + (j * 16 + lo) * 72;
      for (int kc = 0; kc < 2; ++kc)
        bK[j][kc] = frag16(base + kc * 32 + hi * 8, base + kc * 32 + 16 + hi * 8);
    }
    v8f S[2][4];
    for (int i = 0; i < 2; ++i)
      for (int j = 0; j < 4; ++j) {
        v8f s = v8f_zero();
        s = WMMA_BF16(qfr[i][0], bK[j][0], s);
        s = WMMA_BF16(qfr[i][1], bK[j][1], s);
        const int col  = kvbase + j * 16 + lo;
        const int row0 = qbase + i * 16 + hi * 8;
        for (int r = 0; r < 8; ++r) {
          if (col >= N_)            s[r] = NINF_;
          else if (col == row0 + r) s[r] = NEG_;   // LSA diagonal mask
        }
        S[i][j] = s;
      }

    // online softmax (row reductions stay inside the 16-lane half: wave32)
    for (int i = 0; i < 2; ++i)
      for (int r = 0; r < 8; ++r) {
        float mx = fmaxf(fmaxf(S[i][0][r], S[i][1][r]), fmaxf(S[i][2][r], S[i][3][r]));
        mx = fmaxf(mx, __shfl_xor(mx, 1, 32));
        mx = fmaxf(mx, __shfl_xor(mx, 2, 32));
        mx = fmaxf(mx, __shfl_xor(mx, 4, 32));
        mx = fmaxf(mx, __shfl_xor(mx, 8, 32));
        float mnew  = fmaxf(m[i][r], mx);
        float alpha = __expf(m[i][r] - mnew);
        m[i][r] = mnew;
        float rs = 0.f;
        for (int j = 0; j < 4; ++j) {
          float p = __expf(S[i][j][r] - mnew);
          S[i][j][r] = p; rs += p;
        }
        rs += __shfl_xor(rs, 1, 32);
        rs += __shfl_xor(rs, 2, 32);
        rs += __shfl_xor(rs, 4, 32);
        rs += __shfl_xor(rs, 8, 32);
        l[i][r] = l[i][r] * alpha + rs;
        for (int dt = 0; dt < 4; ++dt) O[i][dt][r] *= alpha;
      }

    // P: C-frag layout -> A-frag layout via per-wave LDS (in-order DS per wave)
    unsigned short* pw = ldsP + wave * 32 * 72;
    for (int i = 0; i < 2; ++i)
      for (int j = 0; j < 4; ++j)
        for (int r = 0; r < 8; ++r)
          pw[(i * 16 + hi * 8 + r) * 72 + j * 16 + lo] = f2bf(S[i][j][r]);

    v16bf aP[2][2], bV[4][2];
    for (int i = 0; i < 2; ++i) {
      const unsigned short* base = pw + (i * 16 + lo) * 72;
      for (int kc = 0; kc < 2; ++kc)
        aP[i][kc] = frag16(base + kc * 32 + hi * 8, base + kc * 32 + 16 + hi * 8);
    }
    for (int dt = 0; dt < 4; ++dt) {
      const unsigned short* base = bVlds + (dt * 16 + lo) * 72;
      for (int kc = 0; kc < 2; ++kc)
        bV[dt][kc] = frag16(base + kc * 32 + hi * 8, base + kc * 32 + 16 + hi * 8);
    }
    for (int i = 0; i < 2; ++i)
      for (int dt = 0; dt < 4; ++dt) {
        O[i][dt] = WMMA_BF16(aP[i][0], bV[dt][0], O[i][dt]);
        O[i][dt] = WMMA_BF16(aP[i][1], bV[dt][1], O[i][dt]);
      }
    if (kt + 1 < 10) async_wait();
    __syncthreads();
  }

  // normalize + store to [B, N, H*DH] bf16
  for (int i = 0; i < 2; ++i) {
    int row0 = qbase + i * 16 + hi * 8;
    for (int r = 0; r < 8; ++r) {
      int row = row0 + r;
      if (row < N_) {
        float inv = 1.0f / l[i][r];
        size_t base = ((size_t)(b * N_ + row) * H_ + h) * DH_;
        for (int dt = 0; dt < 4; ++dt)
          aob[base + dt * 16 + lo] = f2bf(O[i][dt][r] * inv);
      }
    }
  }
}

// ---------------------------------------------------------------- out projection
__global__ void __launch_bounds__(256) out_gemm(const unsigned short* __restrict__ aob,
                                                const unsigned short* __restrict__ wot,
                                                const float* __restrict__ bias,
                                                float* __restrict__ out) {
  __shared__ unsigned short ldsA[2 * 128 * 40];
  __shared__ unsigned short ldsB[2 * 128 * 40];
  v8f C[2][4];
  const int mBase = blockIdx.x * 128, nBase = blockIdx.y * 128;
  gemm_mainloop(aob, wot, M_TOK, DIM_, mBase, nBase, ldsA, ldsB, C);

  const int tid = threadIdx.x, wave = tid >> 5, lane = tid & 31;
  const int lo = lane & 15, hi = lane >> 4;
  const int wm = wave >> 1, wn = wave & 1;
  for (int i = 0; i < 2; ++i) {
    int row0 = mBase + wm * 32 + i * 16 + hi * 8;
    for (int j = 0; j < 4; ++j) {
      int col = nBase + wn * 64 + j * 16 + lo;
      float bv = bias[col];
      for (int r = 0; r < 8; ++r) {
        int row = row0 + r;
        if (row < M_TOK) out[(size_t)row * DIM_ + col] = C[i][j][r] + bv;
      }
    }
  }
}

// ---------------------------------------------------------------- launcher
extern "C" void kernel_launch(void* const* d_in, const int* in_sizes, int n_in,
                              void* d_out, int out_size, void* d_ws, size_t ws_size,
                              hipStream_t stream) {
  const float* x     = (const float*)d_in[0];
  const float* W_qkv = (const float*)d_in[1];
  const float* scale = (const float*)d_in[2];
  const float* W_out = (const float*)d_in[3];
  const float* b_out = (const float*)d_in[4];
  float* out = (float*)d_out;

  const int XN    = M_TOK * DIM_;          // 14,180,352
  const int WQN   = DIM_ * 3 * DIM_;       //  1,769,472
  const int WON   = DIM_ * DIM_;           //    589,824
  const int HEADN = B_ * H_ * N_ * DH_;    // 14,180,352
  const int VTN   = B_ * H_ * DH_ * VST_;  // 15,728,640

  unsigned short* ws  = (unsigned short*)d_ws;
  unsigned short* xb  = ws;
  unsigned short* wqt = xb  + XN;
  unsigned short* wot = wqt + WQN;
  unsigned short* qb  = wot + WON;
  unsigned short* kb  = qb  + HEADN;
  unsigned short* vtb = kb  + HEADN;
  unsigned short* aob = vtb + VTN;

  cvt_f32_bf16<<<(XN  + 255) / 256, 256, 0, stream>>>(x, xb, XN);
  transpose_w <<<(WQN + 255) / 256, 256, 0, stream>>>(W_qkv, wqt, DIM_, 3 * DIM_);
  transpose_w <<<(WON + 255) / 256, 256, 0, stream>>>(W_out, wot, DIM_, DIM_);

  qkv_gemm  <<<dim3((M_TOK + 127) / 128, (3 * DIM_) / 128), 256, 0, stream>>>(
      xb, wqt, scale, qb, kb, vtb);
  flash_attn<<<dim3((N_ + 127) / 128, B_ * H_), 128, 0, stream>>>(qb, kb, vtb, aob);
  out_gemm  <<<dim3((M_TOK + 127) / 128, DIM_ / 128), 256, 0, stream>>>(
      aob, wot, b_out, out);
}
